// AVNNType1Conv2d_19902878450223
// MI455X (gfx1250) — matrix operand order repaired and checked by reading, not compiled, not run on hardware
//
#include <hip/hip_runtime.h>

// ---- CDNA5 / gfx1250 vector types -----------------------------------------
typedef __bf16 bf16_t;
typedef __attribute__((ext_vector_type(16))) __bf16 v16bf;  // WMMA A/B (bf16, K=32)
typedef __attribute__((ext_vector_type(8)))  float  v8f;    // WMMA C/D (16x16 f32)
typedef __attribute__((ext_vector_type(2)))  float  f32x2;  // {act_in, car_in} pair

#define Bn   8
#define Cn   128
#define Hn   256
#define Wn   256
#define On   128
#define HWn  (Hn * Wn)
#define KCH  4          // K-chunks of 32 covering C=128
#define NCH  8          // N-chunks of 16 covering O=128
#define NGROUPS (Bn * Hn * (Wn / 16))   // 32768 16-pixel tiles
#define EPSv 1e-6f

__global__ __launch_bounds__(256)
void avnn_type1_conv2d_kernel(const float* __restrict__ x,
                              const float* __restrict__ wx,
                              const float* __restrict__ bx,
                              const float* __restrict__ wy,
                              const float* __restrict__ by,
                              float* __restrict__ out)
{
    // B-fragments for both weight matrices, pre-swizzled to WMMA B fragments in contiguous k order:
    // [mat][kk][oo][lane] -> 16 bf16 (K = (lane>=16 ? 16 : 0) + e), N = lane&15.
    // 2*4*8*32 * 32B = 64 KB LDS (CDNA5 WGP has 320 KB).
    __shared__ v16bf wfrag[2][KCH][NCH][32];

    const int tid  = threadIdx.x;
    const int lane = tid & 31;           // wave32
    const int wave = tid >> 5;
    const int n    = lane & 15;          // N / M position within half-wave
    const int half = lane >> 4;

    // ---- one-time per block: fp32 weights -> bf16 B fragments in LDS -------
    for (int idx = tid; idx < 2 * KCH * NCH * 32; idx += (int)blockDim.x) {
        const int l   = idx & 31;
        const int oo  = (idx >> 5) & 7;
        const int kk  = (idx >> 8) & 3;
        const int mat = idx >> 10;
        const float* __restrict__ wsrc = mat ? wy : wx;
        const int o     = oo * 16 + (l & 15);
        const int cbase = kk * 32 + (l >> 4) * 16;
        v16bf f;
#pragma unroll
        for (int e = 0; e < 16; ++e)
            f[e] = (bf16_t)wsrc[o * Cn + cbase + e];
        wfrag[mat][kk][oo][l] = f;
    }
    __syncthreads();

    // Per-lane biases for each 16-wide output tile (tiny, L2/WGP$-resident).
    float bxv[NCH], byv[NCH];
#pragma unroll
    for (int oo = 0; oo < NCH; ++oo) {
        bxv[oo] = bx[oo * 16 + n];
        byv[oo] = by[oo * 16 + n];
    }

    // ---- tile decode (wave-uniform: EXEC stays all-ones for WMMA) ----------
    const int group = blockIdx.x * (blockDim.x >> 5) + wave;
    if (group >= NGROUPS) return;
    const int gw = group & 15;             // which 16-pixel strip in W
    const int gh = (group >> 4) & (Hn - 1);
    const int gb = group >> 12;
    const int w0 = gw * 16;

    // A-matrix: lane handles pixel m = lane&15; float2 index at c=0.
    const int m       = n;
    const int pixbase = gb * Cn * HWn + gh * Wn + (w0 + m);
    const f32x2* __restrict__ x2 = (const f32x2*)x;

    // ---- load + elementwise derive + downcast into A fragments -------------
    // A fragment k order used here (contiguous): K = 16*(lane>=16) + e (matches B fragments).
    v16bf afrag[KCH], cfrag[KCH];
#pragma unroll
    for (int kk = 0; kk < KCH; ++kk) {
#pragma unroll
        for (int e = 0; e < 16; ++e) {
            const int k = 16 * half + e;
            const int c = kk * 32 + k;
            // one b64 load yields both branches' inputs; NT: pure stream
            const f32x2 v = __builtin_nontemporal_load(&x2[pixbase + c * HWn]);
            const float a = v.x;
            const float carry =
                a * v.y * __builtin_amdgcn_rcpf(__builtin_fabsf(a) + EPSv);
            afrag[kk][e] = (bf16_t)a;
            cfrag[kk][e] = (bf16_t)carry;
        }
    }

    // ---- 8 output tiles x (4 K-chunks x 2 matrices) WMMAs ------------------
    // unroll-by-2 only: keeps accumulators+A fragments under 256 VGPRs so the
    // backend avoids s_set_vgpr_msb churn and ~6 waves/SIMD stay resident to
    // hide HBM latency (this kernel is bandwidth-bound: ~1 GiB @ 23.3 TB/s).
    f32x2* __restrict__ o2 = (f32x2*)out;
#pragma unroll 2
    for (int oo = 0; oo < NCH; ++oo) {
        v8f accA = {};
        v8f accC = {};
#pragma unroll
        for (int kk = 0; kk < KCH; ++kk) {
            const v16bf bwA = wfrag[0][kk][oo][lane];
            accA = __builtin_amdgcn_wmma_f32_16x16x32_bf16(
                false, afrag[kk], false, bwA, (short)0, accA, false, false);
            const v16bf bwC = wfrag[1][kk][oo][lane];
            accC = __builtin_amdgcn_wmma_f32_16x16x32_bf16(
                false, cfrag[kk], false, bwC, (short)0, accC, false, false);
        }
        // D layout: lane -> N = lane&15; VGPR v -> M = v + 8*(lane>=16)
        const int o     = oo * 16 + n;
        const int obase = gb * On * HWn + o * HWn + gh * Wn + w0;
#pragma unroll
        for (int v = 0; v < 8; ++v) {
            const int mrow = v + 8 * half;
            float a = accA[v] + bxv[oo];
            a = a > 0.0f ? a : 0.0f;                 // ReLU (act branch)
            f32x2 r;
            r.x = a;
            r.y = accC[v] + byv[oo];                 // carry branch + bias
            __builtin_nontemporal_store(r, &o2[obase + mrow]);
        }
    }
}

extern "C" void kernel_launch(void* const* d_in, const int* in_sizes, int n_in,
                              void* d_out, int out_size, void* d_ws, size_t ws_size,
                              hipStream_t stream)
{
    const float* x  = (const float*)d_in[0];
    const float* wx = (const float*)d_in[1];
    const float* bx = (const float*)d_in[2];
    const float* wy = (const float*)d_in[3];
    const float* by = (const float*)d_in[4];
    float* out = (float*)d_out;

    // 32768 wave-tiles / 8 waves per 256-thread block = 4096 blocks (exact)
    dim3 grid(NGROUPS / 8), block(256);
    avnn_type1_conv2d_kernel<<<grid, block, 0, stream>>>(x, wx, bx, wy, by, out);
}
